// LaBraM_TimeFilter_SOZ_65309272703228
// MI455X (gfx1250) — matrix operand order repaired and checked by reading, not compiled, not run on hardware
//
#include <hip/hip_runtime.h>
#include <hip/hip_bf16.h>

// ---------------- problem constants ----------------
#define B_      4
#define HEADS_  8
#define L_      1408
#define HD_     64
#define DIM_    512
#define ROWS_   (B_ * L_)        // 5632
#define KEEP_   704              // survivors per row (top half)
#define RTILES_ (ROWS_ / 16)     // 352
#define LTILES_ (L_ / 16)        // 88
#define NPL_    44               // L_/32: row elements per lane

typedef __bf16 bf16_t;
typedef __attribute__((ext_vector_type(16))) __bf16 v16bf;
typedef __attribute__((ext_vector_type(8)))  __bf16 v8bf;
typedef __attribute__((ext_vector_type(8)))  float  v8f;

#if __has_builtin(__builtin_amdgcn_tensor_load_to_lds) && __has_builtin(__builtin_amdgcn_s_wait_tensorcnt)
#define HAVE_TDM 1
typedef unsigned int u32x4 __attribute__((ext_vector_type(4)));
typedef int          i32x4 __attribute__((ext_vector_type(4)));
typedef int          i32x8 __attribute__((ext_vector_type(8)));
#endif

// ---------------- WMMA helpers ----------------
__device__ __forceinline__ v8f wmma_bf16(v16bf a, v16bf b, v8f c) {
  return __builtin_amdgcn_wmma_f32_16x16x32_bf16(
      /*neg_a=*/false, a, /*neg_b=*/false, b,
      /*c_mod=*/(short)0, c, /*reuse_a=*/false, /*reuse_b=*/false);
}

// A-matrix (16x32 bf16): lane = hi*16 + m (m = row). Per lane: one contiguous
// 16-element chunk at K = k0 + hi*16.
__device__ __forceinline__ v16bf load_a_bf16(const bf16_t* row, int k0, int hi) {
  const bf16_t* p = row + k0 + hi * 16;
  v8bf x0 = *(const v8bf*)(p);
  v8bf x1 = *(const v8bf*)(p + 8);
  v16bf r;
#pragma unroll
  for (int i = 0; i < 8; ++i) { r[i] = x0[i]; r[i + 8] = x1[i]; }
  return r;
}

// B-matrix (32x16 bf16): lane = hi*16 + n (n = col). One contiguous 16-element
// chunk at K = k0 + hi*16; "row" = data for column n, contiguous along K.
__device__ __forceinline__ v16bf load_b_bf16(const bf16_t* row, int k0, int hi) {
  return *(const v16bf*)(row + k0 + hi * 16);
}

// monotonic f32 -> sortable u32 key
__device__ __forceinline__ unsigned f32key(float f) {
  unsigned u = __float_as_uint(f);
  return (u & 0x80000000u) ? ~u : (u | 0x80000000u);
}

// ---- stage one contiguous 16 KB (8192 bf16) global tile into LDS offset 0 ----
// TDM path: 1-D tensor descriptor, issued by wave 0 only, tracked by TENSORcnt.
// Requires `l` to be the FIRST (offset-0) LDS object of the kernel.
__device__ __forceinline__ void stage_16k(const bf16_t* __restrict__ g, bf16_t* l, int tid) {
#ifdef HAVE_TDM
  if (tid < 32) {
    unsigned long long ga = (unsigned long long)g;
    u32x4 g0;
    g0[0] = 1u;                                              // count=1, user mode
    g0[1] = 0u;                                              // lds_addr = 0 (first LDS object)
    g0[2] = (unsigned)(ga & 0xffffffffu);                    // global_addr[31:0]
    g0[3] = (unsigned)((ga >> 32) & 0x01ffffffu) | (2u << 30); // global_addr[56:32], type=2
    i32x8 g1 = {};
    g1[0] = 1 << 16;                                         // data_size = 2 bytes, mask=0
    g1[1] = (int)((8192u & 0xffffu) << 16);                  // tensor_dim0[15:0] -> bits 63:48
    g1[2] = (int)(8192u >> 16);                              // tensor_dim0[31:16] -> bits 79:64
    g1[3] = (int)((8192u & 0xffffu) << 16);                  // tile_dim0 -> bits 127:112
    i32x4 z4 = {};
    i32x8 z8 = {};
    __builtin_amdgcn_tensor_load_to_lds(g0, g1, z4, z4, z8, 0);  // 6-arg form
    __builtin_amdgcn_s_wait_tensorcnt(0);
  }
#else
  const v8bf* gs = (const v8bf*)g;
  v8bf* ls = (v8bf*)l;
  for (int i = tid; i < 1024; i += 256) ls[i] = gs[i];
#endif
  __syncthreads();
}

// ---------------- f32 -> bf16 convert ----------------
__global__ void cvt_f32_bf16(const float* __restrict__ in, bf16_t* __restrict__ out, int n) {
  int i = blockIdx.x * 256 + threadIdx.x;
  if (i < n) out[i] = (bf16_t)in[i];
}

// ---------------- K1: qkv = x @ W^T, split to q / k / vT (bf16) ----------------
// All 8 waves of a block share one weight tile (ct uniform: RTILES_ % 8 == 0),
// staged once in LDS via TDM. q,k: [B][H][L][HD]; vT: [B][H][HD][L]; q *= 0.125
__global__ __launch_bounds__(256) void qkv_gemm(const bf16_t* __restrict__ xb,
                                                const bf16_t* __restrict__ wb,
                                                bf16_t* __restrict__ qo,
                                                bf16_t* __restrict__ ko,
                                                bf16_t* __restrict__ vto) {
  __shared__ __align__(16) bf16_t sB[16 * DIM_];       // 16 KB weight tile
  const int tid = threadIdx.x;
  const int wave = tid >> 5, lane = tid & 31;
  const int m = lane & 15, hi = lane >> 4;
  const int ct = (blockIdx.x * 8) / RTILES_;           // uniform in block (0..95)
  const int rt = (blockIdx.x * 8 + wave) % RTILES_;

  stage_16k(wb + (size_t)ct * 16 * DIM_, sB, tid);

  const bf16_t* arow = xb + (size_t)(rt * 16 + m) * DIM_;
  const bf16_t* brow = &sB[m * DIM_];
  // 8 accumulators: same-accumulator WMMA reuse distance 8 -> hazard-free
  v8f acc[8] = {};
#pragma unroll
  for (int u = 0; u < 16; ++u) {
    const int k0 = u * 32;
    acc[u & 7] = wmma_bf16(load_a_bf16(arow, k0, hi), load_b_bf16(brow, k0, hi), acc[u & 7]);
  }
  v8f c = ((acc[0] + acc[1]) + (acc[2] + acc[3])) + ((acc[4] + acc[5]) + (acc[6] + acc[7]));

  const int e0 = ct * 16;
  const int region = e0 >> 9;                          // 0=q 1=k 2=v
  const int h  = (e0 & 511) >> 6;
  const int hd = (e0 & 63) + m;
  const int gr0 = rt * 16;
  const int b  = gr0 / L_;                             // never crosses batch
  const int l0 = gr0 - b * L_;

  if (region == 0) {
    bf16_t* dst = qo + (((size_t)(b * HEADS_ + h) * L_) + l0) * HD_ + hd;
#pragma unroll
    for (int r = 0; r < 8; ++r) dst[(size_t)(r + hi * 8) * HD_] = (bf16_t)(c[r] * 0.125f);
  } else if (region == 1) {
    bf16_t* dst = ko + (((size_t)(b * HEADS_ + h) * L_) + l0) * HD_ + hd;
#pragma unroll
    for (int r = 0; r < 8; ++r) dst[(size_t)(r + hi * 8) * HD_] = (bf16_t)c[r];
  } else {
    bf16_t* dst = vto + ((size_t)(b * HEADS_ + h) * HD_ + hd) * L_ + l0;
#pragma unroll
    for (int r = 0; r < 8; ++r) dst[r + hi * 8] = (bf16_t)c[r];
  }
}

// ---------------- K2: fused scores + bias + top-k drop + softmax + P·V --------
// One block = 16 query rows of one (b,h). 16x1408 f32 scores (90 KB) +
// 16x1408 bf16 probabilities (44 KB) in LDS.
__global__ __launch_bounds__(256) void attn_kernel(const bf16_t* __restrict__ q,
                                                   const bf16_t* __restrict__ kmat,
                                                   const bf16_t* __restrict__ vT,
                                                   const float* __restrict__ region_w,
                                                   const float* __restrict__ region_masks,
                                                   bf16_t* __restrict__ ao) {
  __shared__ __align__(32) float  sS[16 * L_];         // 90112 B (scores, later scratch)
  __shared__ __align__(16) bf16_t sP[16 * L_];         // 45056 B (probabilities)

  const int wave = threadIdx.x >> 5, lane = threadIdx.x & 31;
  const int m = lane & 15, hi = lane >> 4;

  const int tile = blockIdx.x;
  const int lt = tile % LTILES_;
  const int bh = tile / LTILES_;
  const int h = bh & 7, b = bh >> 3;
  const int l0 = lt * 16;

  const bf16_t* qbase = q    + (size_t)bh * L_ * HD_;
  const bf16_t* kbase = kmat + (size_t)bh * L_ * HD_;
  const float w0 = region_w[h * 3 + 0];
  const float w1 = region_w[h * 3 + 1];
  const float w2 = region_w[h * 3 + 2];

  // ---- Phase A: S = (q*scale) K^T + region bias; each wave owns 176 cols ----
  const bf16_t* arow = qbase + (size_t)(l0 + m) * HD_;
  const v16bf a0 = load_a_bf16(arow, 0, hi);
  const v16bf a1 = load_a_bf16(arow, 32, hi);
#pragma unroll
  for (int t = 0; t < 11; ++t) {
    const int n0 = (wave * 11 + t) * 16;
    const bf16_t* brow = kbase + (size_t)(n0 + m) * HD_;
    if (t < 10) __builtin_prefetch(brow + 16 * HD_, 0, 0);   // global_prefetch_b8
    v8f c0 = {}, c1 = {};                                    // independent WMMAs
    c0 = wmma_bf16(a0, load_b_bf16(brow, 0, hi), c0);
    c1 = wmma_bf16(a1, load_b_bf16(brow, 32, hi), c1);
    v8f c = c0 + c1;
    const int col = n0 + m;
#pragma unroll
    for (int r = 0; r < 8; ++r) {
      const int row = r + hi * 8;
      const float* mr = region_masks + (size_t)(l0 + row) * 3 * L_ + col;
      const float bias = w0 * mr[0] + w1 * mr[L_] + w2 * mr[2 * L_];
      sS[row * L_ + col] = c[r] + bias;
    }
  }
  __syncthreads();

  // ---- Phase B: per-row radix-select of 704-th largest + softmax.
  //      Row is loaded ONCE into 44 VGPRs; selection/exp/sum run register-
  //      resident (no LDS re-reads); bf16 result written to sP. ----
#pragma unroll
  for (int rr = 0; rr < 2; ++rr) {
    const int row = wave * 2 + rr;
    const float* S = &sS[row * L_];
    bf16_t* P = &sP[row * L_];

    float v[NPL_];
#pragma unroll
    for (int j = 0; j < NPL_; ++j) v[j] = S[lane + 32 * j];

    unsigned thr = 0u;
    for (int bit = 31; bit >= 0; --bit) {
      const unsigned cand = thr | (1u << bit);
      int cnt = 0;
#pragma unroll
      for (int j = 0; j < NPL_; ++j) cnt += (f32key(v[j]) >= cand) ? 1 : 0;
#pragma unroll
      for (int off = 16; off; off >>= 1) cnt += __shfl_xor(cnt, off, 32);
      if (cnt >= KEEP_) thr = cand;
    }

    float mx = -3.0e38f;
#pragma unroll
    for (int j = 0; j < NPL_; ++j) mx = fmaxf(mx, v[j]);
#pragma unroll
    for (int off = 16; off; off >>= 1) mx = fmaxf(mx, __shfl_xor(mx, off, 32));

    float sum = 0.f;
#pragma unroll
    for (int j = 0; j < NPL_; ++j) {
      const float e = (f32key(v[j]) >= thr) ? __expf(v[j] - mx) : 0.f;
      v[j] = e;
      sum += e;
    }
#pragma unroll
    for (int off = 16; off; off >>= 1) sum += __shfl_xor(sum, off, 32);
    const float inv = 1.0f / sum;

#pragma unroll
    for (int j = 0; j < NPL_; ++j) P[lane + 32 * j] = (bf16_t)(v[j] * inv);
  }
  __syncthreads();

  // ---- Phase C: out = P @ V, all 8 waves: 4 col-tiles x 2-way K split.
  //      Partial accumulators reduced through sS (free after Phase B). ----
  {
    const int nt = wave & 3, kh = wave >> 2;
    const int n0 = nt * 16;
    const bf16_t* vbase = vT + (size_t)bh * HD_ * L_ + (size_t)(n0 + m) * L_;
    const bf16_t* prow  = &sP[m * L_];
    const int kbeg = kh * (L_ / 2);
    v8f acc[8] = {};                                   // 22 WMMAs round-robin
#pragma unroll 1
    for (int u = 0; u < 2; ++u) {                      // 2 x 8 steps
#pragma unroll
      for (int j = 0; j < 8; ++j) {
        const int k0 = kbeg + (u * 8 + j) * 32;
        acc[j] = wmma_bf16(load_a_bf16(prow, k0, hi), load_b_bf16(vbase, k0, hi), acc[j]);
      }
    }
#pragma unroll
    for (int j = 0; j < 6; ++j) {                      // tail 6 steps
      const int k0 = kbeg + (16 + j) * 32;
      acc[j] = wmma_bf16(load_a_bf16(prow, k0, hi), load_b_bf16(vbase, k0, hi), acc[j]);
    }
    v8f c = ((acc[0] + acc[1]) + (acc[2] + acc[3])) + ((acc[4] + acc[5]) + (acc[6] + acc[7]));

    if (kh == 1) ((v8f*)sS)[nt * 32 + lane] = c;
    __syncthreads();
    if (kh == 0) {
      v8f o = ((v8f*)sS)[nt * 32 + lane];
      c = c + o;
      bf16_t* dst = ao + ((size_t)(b * L_ + l0)) * DIM_ + h * HD_ + n0 + m;
#pragma unroll
      for (int r = 0; r < 8; ++r) dst[(size_t)(r + hi * 8) * DIM_] = (bf16_t)c[r];
    }
  }
}

// ---------------- K3: out = attn_out @ proj_w^T + proj_b (f32 out) ------------
__global__ __launch_bounds__(256) void proj_gemm(const bf16_t* __restrict__ ab,
                                                 const bf16_t* __restrict__ wb,
                                                 const float* __restrict__ bias,
                                                 float* __restrict__ out) {
  __shared__ __align__(16) bf16_t sB[16 * DIM_];       // 16 KB weight tile
  const int tid = threadIdx.x;
  const int wave = tid >> 5, lane = tid & 31;
  const int m = lane & 15, hi = lane >> 4;
  const int ct = (blockIdx.x * 8) / RTILES_;           // uniform in block (0..31)
  const int rt = (blockIdx.x * 8 + wave) % RTILES_;

  stage_16k(wb + (size_t)ct * 16 * DIM_, sB, tid);

  const bf16_t* arow = ab + (size_t)(rt * 16 + m) * DIM_;
  const bf16_t* brow = &sB[m * DIM_];
  v8f acc[8] = {};                                     // 8 accumulators
#pragma unroll
  for (int u = 0; u < 16; ++u) {
    const int k0 = u * 32;
    acc[u & 7] = wmma_bf16(load_a_bf16(arow, k0, hi), load_b_bf16(brow, k0, hi), acc[u & 7]);
  }
  v8f c = ((acc[0] + acc[1]) + (acc[2] + acc[3])) + ((acc[4] + acc[5]) + (acc[6] + acc[7]));

  const int d = ct * 16 + m;
  const float bv = bias[d];
  float* dst = out + (size_t)(rt * 16) * DIM_ + d;
#pragma unroll
  for (int r = 0; r < 8; ++r) dst[(size_t)(r + hi * 8) * DIM_] = c[r] + bv;
}

// ---------------- launch ----------------
extern "C" void kernel_launch(void* const* d_in, const int* in_sizes, int n_in,
                              void* d_out, int out_size, void* d_ws, size_t ws_size,
                              hipStream_t stream) {
  (void)in_sizes; (void)n_in; (void)out_size; (void)ws_size;
  const float* x            = (const float*)d_in[0];
  const float* qkv_w        = (const float*)d_in[1];
  const float* proj_w       = (const float*)d_in[2];
  const float* proj_b       = (const float*)d_in[3];
  const float* region_w     = (const float*)d_in[4];
  const float* region_masks = (const float*)d_in[5];

  char* ws = (char*)d_ws;
  const size_t XB   = 0;
  const size_t WQKV = XB   + (size_t)ROWS_ * DIM_ * 2;
  const size_t WPRJ = WQKV + (size_t)3 * DIM_ * DIM_ * 2;
  const size_t QO   = WPRJ + (size_t)DIM_ * DIM_ * 2;
  const size_t KO   = QO   + (size_t)B_ * L_ * DIM_ * 2;
  const size_t VTO  = KO   + (size_t)B_ * L_ * DIM_ * 2;
  const size_t AO   = VTO  + (size_t)B_ * L_ * DIM_ * 2;   // total ~30.9 MB

  bf16_t* xb    = (bf16_t*)(ws + XB);
  bf16_t* wqkvb = (bf16_t*)(ws + WQKV);
  bf16_t* wprjb = (bf16_t*)(ws + WPRJ);
  bf16_t* qo    = (bf16_t*)(ws + QO);
  bf16_t* ko    = (bf16_t*)(ws + KO);
  bf16_t* vto   = (bf16_t*)(ws + VTO);
  bf16_t* ao    = (bf16_t*)(ws + AO);

  const int nx = ROWS_ * DIM_;
  const int nw = 3 * DIM_ * DIM_;
  const int np = DIM_ * DIM_;
  cvt_f32_bf16<<<(nx + 255) / 256, 256, 0, stream>>>(x, xb, nx);
  cvt_f32_bf16<<<(nw + 255) / 256, 256, 0, stream>>>(qkv_w, wqkvb, nw);
  cvt_f32_bf16<<<(np + 255) / 256, 256, 0, stream>>>(proj_w, wprjb, np);

  qkv_gemm<<<(RTILES_ * 96) / 8, 256, 0, stream>>>(xb, wqkvb, qo, ko, vto);          // 4224 blocks
  attn_kernel<<<B_ * HEADS_ * LTILES_, 256, 0, stream>>>(qo, ko, vto,
                                                         region_w, region_masks, ao); // 2816 blocks
  proj_gemm<<<(RTILES_ * 32) / 8, 256, 0, stream>>>(ao, wprjb, proj_b, (float*)d_out); // 1408 blocks
}
